// Block_37203006718445
// MI455X (gfx1250) — matrix operand order repaired and checked by reading, not compiled, not run on hardware
//
#include <hip/hip_runtime.h>

// ---------------------------------------------------------------------------
// Transformer block (pre-norm attn + FFN) for MI455X / gfx1250, wave32 WMMA.
// B=4, T=2048, D=512, H=8, DH=64.  All matmuls via v_wmma_f32_16x16x32_f16.
// ---------------------------------------------------------------------------

typedef __attribute__((ext_vector_type(16))) _Float16 v16h;
typedef __attribute__((ext_vector_type(8)))  _Float16 v8h;
typedef __attribute__((ext_vector_type(8)))  float    v8f;

// ---- WMMA fragment helpers (contiguous K order per lane half in A and B, wave32) -----

// A fragment: 16x32 f16, row-major source, ld in elements.
// lane L: row = L&15, K-base = (L>>4)*16; elems i -> K=kb+i.
// Two contiguous 16B loads.
static __device__ __forceinline__ v16h frag_a(const _Float16* p, int ld) {
  const int lane = threadIdx.x & 31;
  const _Float16* r = p + (size_t)(lane & 15) * ld + ((lane >> 4) << 4);
  v8h lo = *(const v8h*)(r);
  v8h hi = *(const v8h*)(r + 8);
  return __builtin_shufflevector(lo, hi, 0,1,2,3,4,5,6,7,8,9,10,11,12,13,14,15);
}

// B fragment: 32x16 f16 where the source is stored [N][K] row-major
// (i.e. B-transposed storage).  lane L: col = L&15, K-base = (L>>4)*16,
// elems i -> K=kb+i.  Single contiguous 32B load.
static __device__ __forceinline__ v16h frag_bt(const _Float16* p, int ld) {
  const int lane = threadIdx.x & 31;
  return *(const v16h*)(p + (size_t)(lane & 15) * ld + ((lane >> 4) << 4));
}

static __device__ __forceinline__ v8f wmma_f16(v16h a, v16h b, v8f c) {
  // (neg_a, A, neg_b, B, c_mod, C, reuse_a, reuse_b)
  return __builtin_amdgcn_wmma_f32_16x16x32_f16(false, a, false, b,
                                                (short)0, c, false, false);
}

// cross-lane reduce within each 16-lane half (rows live in one half-wave)
static __device__ __forceinline__ float half_max(float t) {
  t = fmaxf(t, __shfl_xor(t, 1));
  t = fmaxf(t, __shfl_xor(t, 2));
  t = fmaxf(t, __shfl_xor(t, 4));
  t = fmaxf(t, __shfl_xor(t, 8));
  return t;
}
static __device__ __forceinline__ float half_sum(float t) {
  t += __shfl_xor(t, 1);
  t += __shfl_xor(t, 2);
  t += __shfl_xor(t, 4);
  t += __shfl_xor(t, 8);
  return t;
}

// ---------------------------------------------------------------------------
// Weight f32 -> f16 transpose-convert: out[n*K + k] = in[k*N + n] (per slice z)
// ---------------------------------------------------------------------------
__global__ __launch_bounds__(256) void transpose_cvt_kernel(
    const float* __restrict__ in, _Float16* __restrict__ out, int K, int N) {
  const size_t slice = (size_t)K * N;
  const float* src = in + slice * blockIdx.z;
  _Float16* dst = out + slice * blockIdx.z;
  int idx = blockIdx.x * 256 + threadIdx.x;
  if (idx < K * N) {
    int kk = idx / N, n = idx - kk * N;
    dst[(size_t)n * K + kk] = (_Float16)src[idx];
  }
}

// ---------------------------------------------------------------------------
// LayerNorm: one block per row of 512, output f16 (WMMA A operand)
// ---------------------------------------------------------------------------
__global__ __launch_bounds__(256) void ln_kernel(
    const float* __restrict__ x, const float* __restrict__ gam,
    const float* __restrict__ bet, _Float16* __restrict__ out) {
  __shared__ float s1[256], s2[256];
  const int row = blockIdx.x, tid = threadIdx.x;
  const float* xr = x + (size_t)row * 512;
  float a = xr[tid], b = xr[tid + 256];
  s1[tid] = a + b;
  s2[tid] = a * a + b * b;
  __syncthreads();
  #pragma unroll
  for (int off = 128; off > 0; off >>= 1) {
    if (tid < off) { s1[tid] += s1[tid + off]; s2[tid] += s2[tid + off]; }
    __syncthreads();
  }
  const float mean = s1[0] * (1.0f / 512.0f);
  const float var  = s2[0] * (1.0f / 512.0f) - mean * mean;
  const float rstd = rsqrtf(var + 1e-5f);
  out[(size_t)row * 512 + tid]       = (_Float16)((a - mean) * rstd * gam[tid] + bet[tid]);
  out[(size_t)row * 512 + tid + 256] = (_Float16)((b - mean) * rstd * gam[tid + 256] + bet[tid + 256]);
}

// ---------------------------------------------------------------------------
// QKV: q/k -> [H*B][T][64] f16 ; v -> transposed [H*B][64][T] f16
// grid: (16, H, 3), block 256 (8 waves, each wave a 64x64 tile)
// ---------------------------------------------------------------------------
__global__ __launch_bounds__(256) void qkv_kernel(
    const _Float16* __restrict__ h16,
    const _Float16* __restrict__ WqT, const _Float16* __restrict__ WkT,
    const _Float16* __restrict__ WvT,
    _Float16* __restrict__ qf, _Float16* __restrict__ kf,
    _Float16* __restrict__ vT) {
  const int lane = threadIdx.x & 31, wave = threadIdx.x >> 5;
  const int which = blockIdx.z;
  const _Float16* Wt = (which == 0) ? WqT : (which == 1) ? WkT : WvT;
  const int head = blockIdx.y;
  const int m0 = blockIdx.x * 512 + wave * 64;
  const _Float16* Wh = Wt + (size_t)head * 64 * 512;   // [64][512] = [N][K]
  v8f acc[4][4] = {};
  for (int k = 0; k < 512; k += 32) {
    v16h a[4], b[4];
    #pragma unroll
    for (int i = 0; i < 4; ++i)
      a[i] = frag_a(h16 + (size_t)(m0 + 16 * i) * 512 + k, 512);
    #pragma unroll
    for (int j = 0; j < 4; ++j)
      b[j] = frag_bt(Wh + (size_t)(16 * j) * 512 + k, 512);
    #pragma unroll
    for (int i = 0; i < 4; ++i)
      #pragma unroll
      for (int j = 0; j < 4; ++j)
        acc[i][j] = wmma_f16(a[i], b[j], acc[i][j]);
  }
  const int cn = lane & 15, g = lane >> 4;
  #pragma unroll
  for (int i = 0; i < 4; ++i)
    #pragma unroll
    for (int j = 0; j < 4; ++j)
      #pragma unroll
      for (int e = 0; e < 8; ++e) {
        const int row = m0 + i * 16 + g * 8 + e;   // b*T + t
        const int bb = row >> 11, tt = row & 2047;
        const int col = j * 16 + cn;               // 0..63
        const size_t hb = (size_t)head * 4 + bb;
        const _Float16 hv = (_Float16)acc[i][j][e];
        if (which == 0)      qf[(hb * 2048 + tt) * 64 + col] = hv;
        else if (which == 1) kf[(hb * 2048 + tt) * 64 + col] = hv;
        else                 vT[(hb * 64 + col) * 2048 + tt] = hv;
      }
}

// ---------------------------------------------------------------------------
// Flash attention, causal.  grid: (T/64, H*B), block 128 (4 waves).
// Each wave: one 16-query tile, streams 32-key chunks.
// Outputs head-concatenated o in f16: [B*T][512]
// ---------------------------------------------------------------------------
__global__ __launch_bounds__(128) void attn_kernel(
    const _Float16* __restrict__ qf, const _Float16* __restrict__ kf,
    const _Float16* __restrict__ vT, _Float16* __restrict__ o16) {
  __shared__ __align__(64) _Float16 lds_p[4][16 * 32];
  const int lane = threadIdx.x & 31, wave = threadIdx.x >> 5;
  const int cn = lane & 15, g = lane >> 4;
  const int hb = blockIdx.y;                // head*B + b
  const int head = hb >> 2, bb = hb & 3;
  const _Float16* qp = qf + (size_t)hb * 2048 * 64;
  const _Float16* kp = kf + (size_t)hb * 2048 * 64;
  const _Float16* vp = vT + (size_t)hb * 64 * 2048;   // [64][2048]
  const int qt = blockIdx.x * 64 + wave * 16;
  const v16h a0 = frag_a(qp + (size_t)qt * 64, 64);        // K 0..31
  const v16h a1 = frag_a(qp + (size_t)qt * 64 + 32, 64);   // K 32..63
  v8f o0 = {}, o1 = {}, o2 = {}, o3 = {};
  float mj[8], lj[8];
  #pragma unroll
  for (int e = 0; e < 8; ++e) { mj[e] = -1e30f; lj[e] = 0.0f; }
  const float scale = 0.04419417382415922f;   // 512^-0.5 (ref scales by D)
  const int qrow = qt + g * 8;
  _Float16* pw = &lds_p[wave][0];

  for (int k0 = 0; k0 <= qt + 15; k0 += 32) {
    // S = q . k^T for two 16-key tiles
    const _Float16* kt0 = kp + (size_t)k0 * 64;
    const _Float16* kt1 = kp + (size_t)(k0 + 16) * 64;
    v8f s0 = {}, s1v = {};
    s0  = wmma_f16(a0, frag_bt(kt0, 64),      s0);
    s0  = wmma_f16(a1, frag_bt(kt0 + 32, 64), s0);
    s1v = wmma_f16(a0, frag_bt(kt1, 64),      s1v);
    s1v = wmma_f16(a1, frag_bt(kt1 + 32, 64), s1v);

    // online softmax update; stage P (f16) to LDS in A-layout source form
    #pragma unroll
    for (int e = 0; e < 8; ++e) {
      const int qr = qrow + e;
      float x0 = s0[e]  * scale; if (k0 + cn > qr)      x0 = -1e30f;
      float x1 = s1v[e] * scale; if (k0 + 16 + cn > qr) x1 = -1e30f;
      const float rm = half_max(fmaxf(x0, x1));
      const float nm = fmaxf(mj[e], rm);
      const float corr = __expf(mj[e] - nm);
      const float p0 = __expf(x0 - nm);
      const float p1 = __expf(x1 - nm);
      lj[e] = lj[e] * corr + half_sum(p0 + p1);
      mj[e] = nm;
      o0[e] *= corr; o1[e] *= corr; o2[e] *= corr; o3[e] *= corr;
      pw[(g * 8 + e) * 32 + cn]      = (_Float16)p0;
      pw[(g * 8 + e) * 32 + 16 + cn] = (_Float16)p1;
    }
    __threadfence_block();                 // order DS stores before DS loads
    const v16h ap = frag_a(pw, 32);        // P as 16x32 A fragment
    // O += P . V  (V transposed: contiguous B fragments)
    o0 = wmma_f16(ap, frag_bt(vp + (size_t)0  * 2048 + k0, 2048), o0);
    o1 = wmma_f16(ap, frag_bt(vp + (size_t)16 * 2048 + k0, 2048), o1);
    o2 = wmma_f16(ap, frag_bt(vp + (size_t)32 * 2048 + k0, 2048), o2);
    o3 = wmma_f16(ap, frag_bt(vp + (size_t)48 * 2048 + k0, 2048), o3);
    __threadfence_block();                 // keep next chunk's stores after loads
  }

  #pragma unroll
  for (int e = 0; e < 8; ++e) {
    const int tt = qrow + e;
    const float inv = 1.0f / lj[e];
    const size_t base = ((size_t)bb * 2048 + tt) * 512 + (size_t)head * 64 + cn;
    o16[base + 0]  = (_Float16)(o0[e] * inv);
    o16[base + 16] = (_Float16)(o1[e] * inv);
    o16[base + 32] = (_Float16)(o2[e] * inv);
    o16[base + 48] = (_Float16)(o3[e] * inv);
  }
}

// ---------------------------------------------------------------------------
// Generic GEMM: C[MxN] = A[MxK](f16) * Bt[NxK](f16) + bias (+res) (relu) ->
// f32 or f16.  grid (M/64, N/256), block 128 (4 waves x 64x64 tile).
// ---------------------------------------------------------------------------
template <bool RELU, bool ADD_RES, bool OUT16>
__global__ __launch_bounds__(128) void gemm_kernel(
    const _Float16* __restrict__ A, const _Float16* __restrict__ Bt,
    const float* __restrict__ bias, const float* __restrict__ res,
    float* __restrict__ Cf, _Float16* __restrict__ Ch,
    int M, int N, int K) {
  const int lane = threadIdx.x & 31, wave = threadIdx.x >> 5;
  const int m0 = blockIdx.x * 64;
  const int n0 = blockIdx.y * 256 + wave * 64;
  v8f acc[4][4] = {};
  for (int k = 0; k < K; k += 32) {
    v16h a[4], b[4];
    #pragma unroll
    for (int i = 0; i < 4; ++i)
      a[i] = frag_a(A + (size_t)(m0 + 16 * i) * K + k, K);
    #pragma unroll
    for (int j = 0; j < 4; ++j)
      b[j] = frag_bt(Bt + (size_t)(n0 + 16 * j) * K + k, K);
    #pragma unroll
    for (int i = 0; i < 4; ++i)
      #pragma unroll
      for (int j = 0; j < 4; ++j)
        acc[i][j] = wmma_f16(a[i], b[j], acc[i][j]);
  }
  const int cn = lane & 15, g = lane >> 4;
  #pragma unroll
  for (int i = 0; i < 4; ++i)
    #pragma unroll
    for (int j = 0; j < 4; ++j) {
      const int col = n0 + j * 16 + cn;
      #pragma unroll
      for (int e = 0; e < 8; ++e) {
        const int row = m0 + i * 16 + g * 8 + e;
        float v = acc[i][j][e] + bias[col];
        if constexpr (ADD_RES) v += res[(size_t)row * N + col];
        if constexpr (RELU)    v = fmaxf(v, 0.0f);
        if constexpr (OUT16)   Ch[(size_t)row * N + col] = (_Float16)v;
        else                   Cf[(size_t)row * N + col] = v;
      }
    }
}

// ---------------------------------------------------------------------------
// Host launcher
// ---------------------------------------------------------------------------
extern "C" void kernel_launch(void* const* d_in, const int* in_sizes, int n_in,
                              void* d_out, int out_size, void* d_ws, size_t ws_size,
                              hipStream_t stream) {
  (void)in_sizes; (void)n_in; (void)out_size; (void)ws_size;
  const float* x     = (const float*)d_in[0];
  const float* ln_g  = (const float*)d_in[1];
  const float* ln_b  = (const float*)d_in[2];
  const float* Wq    = (const float*)d_in[3];
  const float* Wk    = (const float*)d_in[4];
  const float* Wv    = (const float*)d_in[5];
  const float* Wproj = (const float*)d_in[6];
  const float* bproj = (const float*)d_in[7];
  const float* W1    = (const float*)d_in[8];
  const float* b1    = (const float*)d_in[9];
  const float* W2    = (const float*)d_in[10];
  const float* b2    = (const float*)d_in[11];

  // workspace carve (~102 MB total)
  char* p = (char*)d_ws;
  auto take = [&](size_t bytes) {
    char* r = p; p += (bytes + 255) & ~(size_t)255; return r;
  };
  _Float16* h16  = (_Float16*)take(8192ull * 512 * 2);   // LN(x) f16
  _Float16* WqT  = (_Float16*)take(8ull * 64 * 512 * 2);
  _Float16* WkT  = (_Float16*)take(8ull * 64 * 512 * 2);
  _Float16* WvT  = (_Float16*)take(8ull * 64 * 512 * 2);
  _Float16* WpT  = (_Float16*)take(512ull * 512 * 2);
  _Float16* W1T  = (_Float16*)take(2048ull * 512 * 2);
  _Float16* W2T  = (_Float16*)take(512ull * 2048 * 2);
  _Float16* qf   = (_Float16*)take(32ull * 2048 * 64 * 2);
  _Float16* kf   = (_Float16*)take(32ull * 2048 * 64 * 2);
  _Float16* vT   = (_Float16*)take(32ull * 64 * 2048 * 2);
  _Float16* o16  = (_Float16*)take(8192ull * 512 * 2);
  float*    x1   = (float*)   take(8192ull * 512 * 4);   // x + sa
  _Float16* h2   = (_Float16*)take(8192ull * 512 * 2);   // LN(x1) f16
  _Float16* u16  = (_Float16*)take(8192ull * 2048 * 2);  // relu(h2 W1 + b1)

  // weights -> f16 transposed [N][K]
  transpose_cvt_kernel<<<dim3((512 * 64 + 255) / 256, 1, 8), 256, 0, stream>>>(Wq, WqT, 512, 64);
  transpose_cvt_kernel<<<dim3((512 * 64 + 255) / 256, 1, 8), 256, 0, stream>>>(Wk, WkT, 512, 64);
  transpose_cvt_kernel<<<dim3((512 * 64 + 255) / 256, 1, 8), 256, 0, stream>>>(Wv, WvT, 512, 64);
  transpose_cvt_kernel<<<dim3((512 * 512 + 255) / 256, 1, 1), 256, 0, stream>>>(Wproj, WpT, 512, 512);
  transpose_cvt_kernel<<<dim3((512 * 2048 + 255) / 256, 1, 1), 256, 0, stream>>>(W1, W1T, 512, 2048);
  transpose_cvt_kernel<<<dim3((2048 * 512 + 255) / 256, 1, 1), 256, 0, stream>>>(W2, W2T, 2048, 512);

  // h = LN(x)
  ln_kernel<<<8192, 256, 0, stream>>>(x, ln_g, ln_b, h16);
  // q,k,v projections
  qkv_kernel<<<dim3(16, 8, 3), 256, 0, stream>>>(h16, WqT, WkT, WvT, qf, kf, vT);
  // causal flash attention -> o (head-concat, f16)
  attn_kernel<<<dim3(32, 32), 128, 0, stream>>>(qf, kf, vT, o16);
  // x1 = x + o @ Wproj + bproj
  gemm_kernel<false, true, false><<<dim3(128, 2), 128, 0, stream>>>(
      o16, WpT, bproj, x, x1, nullptr, 8192, 512, 512);
  // h2 = LN(x1)  (reference reuses ln1 params)
  ln_kernel<<<8192, 256, 0, stream>>>(x1, ln_g, ln_b, h2);
  // u = relu(h2 @ W1 + b1)
  gemm_kernel<true, false, true><<<dim3(128, 8), 128, 0, stream>>>(
      h2, W1T, b1, nullptr, nullptr, u16, 8192, 2048, 512);
  // out = x1 + u @ W2 + b2
  gemm_kernel<false, true, false><<<dim3(128, 2), 128, 0, stream>>>(
      u16, W2T, b2, x1, (float*)d_out, nullptr, 8192, 512, 2048);
}
